// CrossAttention_27384711479999
// MI455X (gfx1250) — matrix operand order repaired and checked by reading, not compiled, not run on hardware
//
#include <hip/hip_runtime.h>
#include <hip/hip_bf16.h>

typedef __attribute__((ext_vector_type(16))) _Float16 v16h;
typedef __attribute__((ext_vector_type(8)))  _Float16 v8h;
typedef __attribute__((ext_vector_type(4)))  _Float16 v4h;
typedef __attribute__((ext_vector_type(8)))  float    v8f;
typedef int v4i __attribute__((__vector_size__(16)));   // matches builtin param

#define N_HEAD 16
#define BB 8
#define TT 1024
#define SS 1024
#define CC 1024
#define DD 64

// ---------------------------------------------------------------------------
// WMMA helper (CDNA5 gfx1250, wave32). D = A(16x32 f16) * B(32x16 f16) + C(f32)
// ---------------------------------------------------------------------------
__device__ inline v8f wmma_f16(v16h a, v16h b, v8f c) {
  return __builtin_amdgcn_wmma_f32_16x16x32_f16(
      /*neg_a=*/false, a, /*neg_b=*/false, b,
      /*c_mod=*/(short)0, c, /*reuse_a=*/false, /*reuse_b=*/false);
}

// A fragment, 16x32, row-major source (ld in halves).
// Lane L<16: row L, K = {0..7, 16..23}; lane L+16: row L, K = {8..15, 24..31}.
__device__ inline v16h frag_a(const _Float16* __restrict__ p, int ld) {
  int lane = threadIdx.x & 31;
  int row  = lane & 15;
  int koff = (lane >> 4) * 8;
  const _Float16* r = p + (size_t)row * ld + koff;
  union { v16h v; v8h h[2]; } u;
  u.h[0] = *(const v8h*)(r);
  u.h[1] = *(const v8h*)(r + 16);
  return u.v;
}

// B fragment, 32x16, source stored N-major (row = N, col = K, ld in halves),
// i.e. B[k][n] = src[n][k]. Lane L<16: col L, K = {0..7, 16..23}; lane L+16: col L,
// K = {8..15, 24..31}.
__device__ inline v16h frag_b(const _Float16* __restrict__ p, int ld) {
  int lane = threadIdx.x & 31;
  int col  = lane & 15;
  int koff = (lane >> 4) * 8;
  const _Float16* r = p + (size_t)col * ld + koff;
  union { v16h v; v8h h[2]; } u;
  u.h[0] = *(const v8h*)(r);
  u.h[1] = *(const v8h*)(r + 16);
  return u.v;
}

// B fragment from an LDS tile with row stride 64 halves (64-wide K chunk);
// klo selects the 32-wide K sub-chunk (0 or 32).
__device__ inline v16h frag_b_lds64(const _Float16* p, int klo) {
  int lane = threadIdx.x & 31;
  int col  = lane & 15;
  int koff = klo + (lane >> 4) * 8;
  const _Float16* r = p + col * 64 + koff;
  union { v16h v; v8h h[2]; } u;
  u.h[0] = *(const v8h*)(r);
  u.h[1] = *(const v8h*)(r + 16);
  return u.v;
}

// ---------------------------------------------------------------------------
// gfx1250 async global->LDS copy (ASYNCcnt path), with inline-asm fallback.
// Each active lane moves 16 bytes from its global address to its LDS address.
// Builtin signature (probe-confirmed): (global v4i*, local v4i*, imm, imm).
// ---------------------------------------------------------------------------
__device__ inline void async_b128(const _Float16* gsrc, _Float16* ldst) {
#if __has_builtin(__builtin_amdgcn_global_load_async_to_lds_b128)
  __builtin_amdgcn_global_load_async_to_lds_b128(
      (__attribute__((address_space(1))) v4i*)gsrc,
      (__attribute__((address_space(3))) v4i*)ldst, 0, 0);
#else
  unsigned lofs = (unsigned)(uintptr_t)(__attribute__((address_space(3))) _Float16*)ldst;
  asm volatile("global_load_async_to_lds_b128 %0, %1, off"
               :: "v"(lofs), "v"((unsigned long long)(uintptr_t)gsrc)
               : "memory");
#endif
}

__device__ inline void wait_async0() {
#if __has_builtin(__builtin_amdgcn_s_wait_asynccnt)
  __builtin_amdgcn_s_wait_asynccnt(0);
#else
  asm volatile("s_wait_asynccnt 0" ::: "memory");
#endif
}

// 16-lane (half-wave) row reductions; rows live across lanes 0..15 / 16..31.
__device__ inline float rowmax16(float v) {
  v = fmaxf(v, __shfl_xor(v, 1, 32));
  v = fmaxf(v, __shfl_xor(v, 2, 32));
  v = fmaxf(v, __shfl_xor(v, 4, 32));
  v = fmaxf(v, __shfl_xor(v, 8, 32));
  return v;
}
__device__ inline float rowsum16(float v) {
  v += __shfl_xor(v, 1, 32);
  v += __shfl_xor(v, 2, 32);
  v += __shfl_xor(v, 4, 32);
  v += __shfl_xor(v, 8, 32);
  return v;
}

// ---------------------------------------------------------------------------
// f32 -> f16 conversion (vectorized, grid-stride)
// ---------------------------------------------------------------------------
__global__ void cvt_f32_f16(const float4* __restrict__ s, v4h* __restrict__ d, int n4) {
  int i  = blockIdx.x * blockDim.x + threadIdx.x;
  int st = gridDim.x * blockDim.x;
  for (; i < n4; i += st) {
    float4 f = s[i];
    v4h h;
    h[0] = (_Float16)f.x; h[1] = (_Float16)f.y;
    h[2] = (_Float16)f.z; h[3] = (_Float16)f.w;
    d[i] = h;
  }
}

// ---------------------------------------------------------------------------
// Shared GEMM core: per wave 16 rows x 64 cols, K accumulated in 64-chunks.
// B tile (64 N-rows x 64 K-halves = 8KB) staged into LDS with async copies,
// double buffered (16 barriers/K-loop instead of 32); A fragments double
// buffered in registers. Block = (32,8): 8 waves, 128 M-rows, shared B tile.
// ---------------------------------------------------------------------------
__device__ inline void stage_btile64(const _Float16* __restrict__ W, int n0, int k0,
                                     int K, _Float16* buf) {
  int tid = threadIdx.y * 32 + threadIdx.x;   // 0..255; 512 16B-chunks total
#pragma unroll
  for (int c = 0; c < 2; ++c) {
    int idx = tid + c * 256;                  // 0..511
    int row = idx >> 3;                       // 0..63
    int seg = idx & 7;                        // 8 x 8 halves = 128B per row
    async_b128(W + (size_t)(n0 + row) * K + k0 + seg * 8, buf + row * 64 + seg * 8);
  }
}

__device__ inline void gemm_core(const _Float16* __restrict__ A,
                                 const _Float16* __restrict__ W,
                                 int m0, int n0, int K,
                                 _Float16* bsm /* [2][64*64] */, v8f acc[4]) {
  stage_btile64(W, n0, 0, K, bsm);
  v16h a_cur0 = frag_a(A + (size_t)m0 * K, K);
  v16h a_cur1 = frag_a(A + (size_t)m0 * K + 32, K);
  wait_async0();
  __syncthreads();
  int cur = 0;
  for (int k0 = 0; k0 < K; k0 += 64) {
    bool has_next = (k0 + 64) < K;
    v16h a_nxt0, a_nxt1;
    if (has_next) {
      stage_btile64(W, n0, k0 + 64, K, bsm + (cur ^ 1) * (64 * 64));  // in flight
      a_nxt0 = frag_a(A + (size_t)m0 * K + (k0 + 64), K);             // in flight
      a_nxt1 = frag_a(A + (size_t)m0 * K + (k0 + 96), K);
    }
    const _Float16* bb = bsm + cur * (64 * 64);
    // Fetch all 4 fragments of the low K-half first, then WMMA: the DS pipe
    // runs ahead so waits are partial instead of dscnt==0 per WMMA.
    v16h b0[4];
#pragma unroll
    for (int j = 0; j < 4; ++j) b0[j] = frag_b_lds64(bb + (16 * j) * 64, 0);
#pragma unroll
    for (int j = 0; j < 4; ++j) acc[j] = wmma_f16(a_cur0, b0[j], acc[j]);
    v16h b1[4];
#pragma unroll
    for (int j = 0; j < 4; ++j) b1[j] = frag_b_lds64(bb + (16 * j) * 64, 32);
#pragma unroll
    for (int j = 0; j < 4; ++j) acc[j] = wmma_f16(a_cur1, b1[j], acc[j]);
    if (has_next) {
      a_cur0 = a_nxt0;
      a_cur1 = a_nxt1;
      wait_async0();
    }
    __syncthreads();
    cur ^= 1;
  }
}

// ---------------------------------------------------------------------------
// GEMM: q = x @ Wc^T + bc  -> q_h[b][h][t][dd]  (f16)
// ---------------------------------------------------------------------------
__global__ void gemm_q(const _Float16* __restrict__ A, const _Float16* __restrict__ W,
                       const float* __restrict__ bias, _Float16* __restrict__ qh) {
  __shared__ _Float16 bsm[2][64 * 64];
  int m0 = (blockIdx.y * blockDim.y + threadIdx.y) * 16;
  int n0 = blockIdx.x * 64;
  v8f acc[4] = {};
  gemm_core(A, W, m0, n0, CC, &bsm[0][0], acc);

  int lane  = threadIdx.x & 31;
  int ncol  = lane & 15;
  int rbase = (lane >> 4) * 8;
#pragma unroll
  for (int j = 0; j < 4; ++j) {
    int n = n0 + 16 * j + ncol;
    int h = n >> 6, dd = n & 63;
    float bv = bias[n];
#pragma unroll
    for (int r = 0; r < 8; ++r) {
      int m = m0 + rbase + r;
      int b = m >> 10, t = m & 1023;
      qh[(((size_t)(b * N_HEAD + h)) * TT + t) * DD + dd] = (_Float16)(acc[j][r] + bv);
    }
  }
}

// ---------------------------------------------------------------------------
// GEMM: kv = feature @ Wf^T + bf.
// n < C  -> k_h[b][h][s][dd]   ; n >= C -> vT_h[b][h][dd][s] (transposed)
// ---------------------------------------------------------------------------
__global__ void gemm_kv(const _Float16* __restrict__ A, const _Float16* __restrict__ W,
                        const float* __restrict__ bias,
                        _Float16* __restrict__ kh, _Float16* __restrict__ vT) {
  __shared__ _Float16 bsm[2][64 * 64];
  int m0 = (blockIdx.y * blockDim.y + threadIdx.y) * 16;
  int n0 = blockIdx.x * 64;
  v8f acc[4] = {};
  gemm_core(A, W, m0, n0, CC, &bsm[0][0], acc);

  int lane  = threadIdx.x & 31;
  int ncol  = lane & 15;
  int rbase = (lane >> 4) * 8;
#pragma unroll
  for (int j = 0; j < 4; ++j) {
    int n = n0 + 16 * j + ncol;
    float bv = bias[n];
#pragma unroll
    for (int r = 0; r < 8; ++r) {
      int m = m0 + rbase + r;
      int b = m >> 10, s = m & 1023;
      float val = acc[j][r] + bv;
      if (n < CC) {
        int h = n >> 6, dd = n & 63;
        kh[(((size_t)(b * N_HEAD + h)) * SS + s) * DD + dd] = (_Float16)val;
      } else {
        int nn = n - CC;
        int h = nn >> 6, dd = nn & 63;
        vT[(((size_t)(b * N_HEAD + h)) * DD + dd) * SS + s] = (_Float16)val;
      }
    }
  }
}

// ---------------------------------------------------------------------------
// Causal flash attention. One wave per (b, h, 16-row q tile); S in 32-wide
// tiles (K=32 for P@V WMMA). All K/V fragments for the tile are issued at the
// loop top so V loads overlap score WMMAs + softmax VALU work.
// ---------------------------------------------------------------------------
__global__ void attn(const _Float16* __restrict__ Q, const _Float16* __restrict__ Kmat,
                     const _Float16* __restrict__ Vt, _Float16* __restrict__ O) {
  int b  = blockIdx.z;
  int h  = blockIdx.y;
  int t0 = (blockIdx.x * blockDim.y + threadIdx.y) * 16;

  int lane  = threadIdx.x & 31;
  int ncol  = lane & 15;
  int hi    = lane >> 4;
  int rbase = hi * 8;

  const _Float16* qp = Q    + (((size_t)(b * N_HEAD + h)) * TT + t0) * DD;
  const _Float16* kp = Kmat + ((size_t)(b * N_HEAD + h)) * SS * DD;
  const _Float16* vp = Vt   + ((size_t)(b * N_HEAD + h)) * DD * SS;

  v16h qa0 = frag_a(qp, DD);        // K(d) = 0..31
  v16h qa1 = frag_a(qp + 32, DD);   // K(d) = 32..63

  float mrow[8], lrow[8];
  v8f o[4] = {};
#pragma unroll
  for (int r = 0; r < 8; ++r) { mrow[r] = -__builtin_inff(); lrow[r] = 0.0f; }

  __shared__ _Float16 plds[4][16 * 40];   // per-wave P staging, stride 40 halves
  _Float16* pl = plds[threadIdx.y];

  const float scale = 0.125f;             // 1/sqrt(64)

  for (int s0 = 0; s0 <= t0; s0 += 32) {
    // ---- issue ALL fragment loads for this tile up front (overlap w/ compute)
    v16h kb0 = frag_b(kp + (size_t)s0 * DD, DD);
    v16h kb1 = frag_b(kp + (size_t)s0 * DD + 32, DD);
    v16h kb2 = frag_b(kp + (size_t)(s0 + 16) * DD, DD);
    v16h kb3 = frag_b(kp + (size_t)(s0 + 16) * DD + 32, DD);
    v16h vb[4];
#pragma unroll
    for (int j = 0; j < 4; ++j) vb[j] = frag_b(vp + (size_t)(j * 16) * SS + s0, SS);

    // prefetch next K tile (4KB contiguous) into cache
    if (s0 + 32 <= t0) {
      __builtin_prefetch(kp + (size_t)(s0 + 32) * DD, 0, 0);
      __builtin_prefetch(kp + (size_t)(s0 + 48) * DD, 0, 0);
    }

    // ---- scores: 16 x 32 tile = two 16x16 WMMA accumulators
    v8f sA = {}, sB = {};
    sA = wmma_f16(qa0, kb0, sA);
    sA = wmma_f16(qa1, kb1, sA);
    sB = wmma_f16(qa0, kb2, sB);
    sB = wmma_f16(qa1, kb3, sB);

    // ---- causal mask + online softmax (per row r held by this lane)
#pragma unroll
    for (int r = 0; r < 8; ++r) {
      int t  = t0 + rbase + r;
      int sa = s0 + ncol;
      int sb = s0 + 16 + ncol;
      float xa = (sa <= t) ? sA[r] * scale : -__builtin_inff();
      float xb = (sb <= t) ? sB[r] * scale : -__builtin_inff();
      float mx   = rowmax16(fmaxf(xa, xb));
      float mnew = fmaxf(mrow[r], mx);
      float corr = __expf(mrow[r] - mnew);
      float pa   = __expf(xa - mnew);
      float pb   = __expf(xb - mnew);
      float rs   = rowsum16(pa + pb);
      lrow[r] = lrow[r] * corr + rs;
      mrow[r] = mnew;
#pragma unroll
      for (int j = 0; j < 4; ++j) o[j][r] *= corr;
      int row = rbase + r;
      pl[row * 40 + ncol]      = (_Float16)pa;
      pl[row * 40 + 16 + ncol] = (_Float16)pb;
    }
    __builtin_amdgcn_wave_barrier();   // same-wave LDS ops are in-order (DScnt)

    // ---- reload P in A-fragment layout (16x32)
    v16h pfrag;
    {
      int prow = ncol;
      int pk   = hi * 8;
      union { v16h v; v8h h2[2]; } u;
      u.h2[0] = *(const v8h*)(pl + prow * 40 + pk);
      u.h2[1] = *(const v8h*)(pl + prow * 40 + pk + 16);
      pfrag = u.v;
    }
    __builtin_amdgcn_wave_barrier();

    // ---- O(16x64) += P(16x32) @ V(32x64)
#pragma unroll
    for (int j = 0; j < 4; ++j) o[j] = wmma_f16(pfrag, vb[j], o[j]);
  }

  // ---- normalize and store as [B][T][C] with C index = h*64 + d
  _Float16* op = O + ((size_t)b * TT + t0) * CC + h * DD;
#pragma unroll
  for (int r = 0; r < 8; ++r) {
    float inv = 1.0f / lrow[r];
    int m = rbase + r;
#pragma unroll
    for (int j = 0; j < 4; ++j) {
      op[(size_t)m * CC + j * 16 + ncol] = (_Float16)(o[j][r] * inv);
    }
  }
}

// ---------------------------------------------------------------------------
// GEMM: out = attout @ Wp^T + bp   (f32 output, row-major [B*T, C])
// ---------------------------------------------------------------------------
__global__ void gemm_proj(const _Float16* __restrict__ A, const _Float16* __restrict__ W,
                          const float* __restrict__ bias, float* __restrict__ out) {
  __shared__ _Float16 bsm[2][64 * 64];
  int m0 = (blockIdx.y * blockDim.y + threadIdx.y) * 16;
  int n0 = blockIdx.x * 64;
  v8f acc[4] = {};
  gemm_core(A, W, m0, n0, CC, &bsm[0][0], acc);

  int lane  = threadIdx.x & 31;
  int ncol  = lane & 15;
  int rbase = (lane >> 4) * 8;
#pragma unroll
  for (int j = 0; j < 4; ++j) {
    int n = n0 + 16 * j + ncol;
    float bv = bias[n];
#pragma unroll
    for (int r = 0; r < 8; ++r) {
      int m = m0 + rbase + r;
      out[(size_t)m * CC + n] = acc[j][r] + bv;
    }
  }
}

// ---------------------------------------------------------------------------
// Launch
// ---------------------------------------------------------------------------
extern "C" void kernel_launch(void* const* d_in, const int* in_sizes, int n_in,
                              void* d_out, int out_size, void* d_ws, size_t ws_size,
                              hipStream_t stream) {
  const float* x  = (const float*)d_in[0];   // [B,T,C]
  const float* ft = (const float*)d_in[1];   // [B,S,C]
  const float* Wc = (const float*)d_in[2];   // [C,C]
  const float* bc = (const float*)d_in[3];   // [C]
  const float* Wf = (const float*)d_in[4];   // [2C,C]
  const float* bf = (const float*)d_in[5];   // [2C]
  const float* Wp = (const float*)d_in[6];   // [C,C]
  const float* bp = (const float*)d_in[7];   // [C]
  float* out = (float*)d_out;

  const size_t nX  = (size_t)BB * TT * CC;   // 8M
  const size_t nWc = (size_t)CC * CC;        // 1M
  const size_t nWf = (size_t)2 * CC * CC;    // 2M
  const size_t nWp = (size_t)CC * CC;        // 1M

  char* ws = (char*)d_ws;
  size_t off = 0;
  auto salloc = [&](size_t bytes) -> void* {
    void* p = ws + off;
    off += (bytes + 255) & ~(size_t)255;
    return p;
  };
  _Float16* x_h  = (_Float16*)salloc(nX * 2);    // 16MB
  _Float16* f_h  = (_Float16*)salloc(nX * 2);    // 16MB
  _Float16* Wc_h = (_Float16*)salloc(nWc * 2);   // 2MB
  _Float16* Wf_h = (_Float16*)salloc(nWf * 2);   // 4MB
  _Float16* Wp_h = (_Float16*)salloc(nWp * 2);   // 2MB
  _Float16* q_h  = (_Float16*)salloc(nX * 2);    // 16MB [B,H,T,d]
  _Float16* k_h  = (_Float16*)salloc(nX * 2);    // 16MB [B,H,S,d]
  _Float16* vT_h = (_Float16*)salloc(nX * 2);    // 16MB [B,H,d,S]
  _Float16* ao_h = (_Float16*)salloc(nX * 2);    // 16MB [B,T,C]
  (void)ws_size;

  // 1) f32 -> f16 staging
  cvt_f32_f16<<<2048, 256, 0, stream>>>((const float4*)x,  (v4h*)x_h,  (int)(nX  / 4));
  cvt_f32_f16<<<2048, 256, 0, stream>>>((const float4*)ft, (v4h*)f_h,  (int)(nX  / 4));
  cvt_f32_f16<<<512,  256, 0, stream>>>((const float4*)Wc, (v4h*)Wc_h, (int)(nWc / 4));
  cvt_f32_f16<<<512,  256, 0, stream>>>((const float4*)Wf, (v4h*)Wf_h, (int)(nWf / 4));
  cvt_f32_f16<<<512,  256, 0, stream>>>((const float4*)Wp, (v4h*)Wp_h, (int)(nWp / 4));

  // 2) q projection: M=8192, N=1024
  gemm_q<<<dim3(CC / 64, (BB * TT) / 128), dim3(32, 8), 0, stream>>>(x_h, Wc_h, bc, q_h);

  // 3) kv projection: M=8192, N=2048 (k + transposed v)
  gemm_kv<<<dim3(2 * CC / 64, (BB * SS) / 128), dim3(32, 8), 0, stream>>>(f_h, Wf_h, bf, k_h, vT_h);

  // 4) causal flash attention: grid (T/64, H, B), 4 waves/block
  attn<<<dim3(TT / 64, N_HEAD, BB), dim3(32, 4), 0, stream>>>(q_h, k_h, vT_h, ao_h);

  // 5) output projection: M=8192, N=1024, f32 out
  gemm_proj<<<dim3(CC / 64, (BB * TT) / 128), dim3(32, 8), 0, stream>>>(ao_h, Wp_h, bp, out);
}
